// RNN_5188320494079
// MI455X (gfx1250) — matrix operand order repaired and checked by reading, not compiled, not run on hardware
//
#include <hip/hip_runtime.h>
#include <hip/hip_bf16.h>
#include <math.h>

#define NB 64      // batch
#define TT 1024    // timesteps
#define DD 256     // input dim
#define HH 256     // hidden dim

typedef __attribute__((ext_vector_type(16))) __bf16 v16bf;
typedef __attribute__((ext_vector_type(8)))  __bf16 v8bf;
typedef __attribute__((ext_vector_type(8)))  float  v8f;
typedef unsigned int u32x4 __attribute__((ext_vector_type(4)));
typedef int          i32x4 __attribute__((ext_vector_type(4)));
typedef int          i32x8 __attribute__((ext_vector_type(8)));

static __device__ __forceinline__ __bf16 f2bf(float f) { return (__bf16)f; }

// ---------------------------------------------------------------------------
// Kernel 0: convert + transpose weights to bf16 in workspace.
//   WxT[h][d] = bf16(Wx[d][h]),  WhT[n][k] = bf16(Wh[k][n])
// ---------------------------------------------------------------------------
__global__ void prep_weights_kernel(const float* __restrict__ Wx,
                                    const float* __restrict__ Wh,
                                    unsigned short* __restrict__ WxT_u,
                                    unsigned short* __restrict__ WhT_u) {
    __bf16* WxT = (__bf16*)WxT_u;
    __bf16* WhT = (__bf16*)WhT_u;
    int i = blockIdx.x * blockDim.x + threadIdx.x;   // 0 .. 65535
    if (i < DD * HH) {
        int d = i / HH;      // row of Wx / Wh
        int h = i % HH;      // col of Wx / Wh (coalesced read)
        WxT[(size_t)h * DD + d] = f2bf(Wx[i]);
        WhT[(size_t)h * HH + d] = f2bf(Wh[i]);
    }
}

// ---------------------------------------------------------------------------
// Kernel 1: xW = x @ Wx + b  written into d_out  (out is [N*T, H] fp32).
// Block: 256 threads = 8 waves. Block tile: 128 rows x 64 cols.
// Wave tile: 16 rows x 64 cols = 4 WMMA tiles; K looped in steps of 32.
// ---------------------------------------------------------------------------
__global__ __launch_bounds__(256)
void xw_gemm_kernel(const float* __restrict__ x,
                    const unsigned short* __restrict__ WxT_u,
                    const float* __restrict__ b,
                    float* __restrict__ out) {
    const __bf16* WxT = (const __bf16*)WxT_u;
    const int lane = threadIdx.x & 31;
    const int wave = threadIdx.x >> 5;
    const int l15  = lane & 15;
    const int hs   = lane >> 4;                 // half-wave select (0/1)

    const int m0    = blockIdx.x * 128 + wave * 16;   // row tile base
    const int nbase = blockIdx.y * 64;                // col tile base

    const int arow = m0 + l15;
    const float* xrow = x + (size_t)arow * DD;

    v8f acc[4];
#pragma unroll
    for (int q = 0; q < 4; ++q)
#pragma unroll
        for (int r = 0; r < 8; ++r) acc[q][r] = 0.0f;

#pragma unroll
    for (int k0 = 0; k0 < DD; k0 += 32) {
        // ---- A fragment (16x32 bf16): lanes 0-15 hold K=k0..k0+15,
        //      lanes 16-31 hold K=k0+16..k0+31
        const int kbase = k0 + (hs ? 16 : 0);
        v8f flo = *(const v8f*)(xrow + kbase);
        v8f fhi = *(const v8f*)(xrow + kbase + 8);
        if (k0 + 64 < DD) __builtin_prefetch(xrow + k0 + 64, 0, 3);
        v16bf a;
#pragma unroll
        for (int i = 0; i < 8; ++i) { a[i] = f2bf(flo[i]); a[8 + i] = f2bf(fhi[i]); }

        // ---- B fragments: lane = column; lanes 0-15 K=k0..k0+15,
        //      lanes 16-31 K=k0+16..k0+31 -> 32B contiguous from WxT
        const int kk = k0 + (hs ? 16 : 0);
#pragma unroll
        for (int q = 0; q < 4; ++q) {
            const int col = nbase + q * 16 + l15;
            v16bf bf = *(const v16bf*)(WxT + (size_t)col * DD + kk);
            acc[q] = __builtin_amdgcn_wmma_f32_16x16x32_bf16(
                false, a, false, bf, (short)0, acc[q], false, false);
        }
    }

    // Epilogue: + bias, store fp32 C tiles (VGPR r <-> row r (+8 upper half))
#pragma unroll
    for (int q = 0; q < 4; ++q) {
        const int col = nbase + q * 16 + l15;
        const float bv = b[col];
#pragma unroll
        for (int r = 0; r < 8; ++r) {
            const int row = m0 + r + (hs ? 8 : 0);
            out[(size_t)row * HH + col] = acc[q][r] + bv;
        }
    }
}

// ---------------------------------------------------------------------------
// Kernel 2: sequential scan, single workgroup (1024 threads = 32 waves on
// one WGP). WhT (128KB bf16, TDM-staged) + double-buffered h (2x32KB bf16)
// live in the 320KB CDNA5 LDS. Each step: h_t = tanh(xw_t + h_{t-1} @ Wh),
// in-place on d_out (xw_t read before h_t overwrite). xw(t+1) loads are
// software-pipelined behind the WMMA chain of step t.
// ---------------------------------------------------------------------------
__global__ __launch_bounds__(1024)
void rnn_scan_kernel(const float* __restrict__ h0,
                     const unsigned short* __restrict__ WhT_u,
                     float* __restrict__ out) {
    extern __shared__ __align__(32) char smem[];
    __bf16* whs = (__bf16*)smem;                            // [HH][HH] 128KB
    __bf16* hb  = (__bf16*)(smem + (size_t)HH * HH * 2);    // [2][NB][HH] 64KB

    const int tid = threadIdx.x;

    // ---- Stage WhT -> LDS.
#if __has_builtin(__builtin_amdgcn_tensor_load_to_lds) && \
    __has_builtin(__builtin_amdgcn_s_wait_tensorcnt)
    // Tensor Data Mover: one DMA of the whole 128KB block, issued by wave 0.
    // D# per ISA 8.3/8.4: 2D tensor, data_size=8B, tile 16384x1 (contiguous).
    if (tid < 32) {
        unsigned lds_off =
            (unsigned)(unsigned long long)(__attribute__((address_space(3))) char*)smem;
        unsigned long long ga = (unsigned long long)WhT_u;
        u32x4 g0;
        g0.x = 1u;                                    // count=1 (valid D#)
        g0.y = lds_off;                               // lds_addr [63:32]
        g0.z = (unsigned)ga;                          // global_addr [95:64]
        g0.w = (unsigned)((ga >> 32) & 0x01FFFFFFu)   // global_addr [120:96]
             | 0x80000000u;                           // type=2 ("image") [127:126]
        i32x8 g1;
        g1[0] = (int)(3u << 16);      // workgroup_mask=0, data_size=3 (8B)
        g1[1] = (int)0x40000000;      // tensor_dim0[15:0]=0x4000 in [63:48]
        g1[2] = (int)0x00010000;      // tensor_dim0[31:16]=0; tensor_dim1[15:0]=1
        g1[3] = (int)0x40000000;      // tensor_dim1 hi=0; tile_dim0=0x4000 [127:112]
        g1[4] = 1;                    // tile_dim1=1; tile_dim2=0
        g1[5] = 0x4000;               // tensor_dim0_stride[31:0]
        g1[6] = 0;                    // stride hi / dim1_stride lo
        g1[7] = 0;
        i32x4 z4 = {0, 0, 0, 0};
#if __clang_major__ >= 23
        i32x8 z8 = {0, 0, 0, 0, 0, 0, 0, 0};
        __builtin_amdgcn_tensor_load_to_lds(g0, g1, z4, z4, z8, 0);
#else
        __builtin_amdgcn_tensor_load_to_lds(g0, g1, z4, z4, 0);
#endif
        __builtin_amdgcn_s_wait_tensorcnt(0);
    }
#else
    {
        const uint4* src = (const uint4*)WhT_u;
        uint4* dst = (uint4*)whs;
        const int nchunks = (HH * HH * 2) / 16;             // 8192
        for (int i = tid; i < nchunks; i += 1024) dst[i] = src[i];
    }
#endif
    // h0 (fp32) -> bf16 into buffer 0
    for (int i = tid; i < NB * HH; i += 1024) hb[i] = f2bf(h0[i]);
    __syncthreads();

    const int lane = tid & 31;
    const int wave = tid >> 5;
    const int l15  = lane & 15;
    const int hs   = lane >> 4;
    const int m0   = (wave & 3) * 16;       // row-tile base (batch dim)
    const int np   = wave >> 2;             // col-pair index 0..7
    const int col0 = np * 32 + l15;
    const int col1 = np * 32 + 16 + l15;
    const int arow = m0 + l15;

    // Preload xw(t=0)
    v8f nxt0, nxt1;
#pragma unroll
    for (int r = 0; r < 8; ++r) {
        const int row = m0 + r + (hs ? 8 : 0);
        const size_t base = (size_t)row * TT * HH;
        nxt0[r] = out[base + col0];
        nxt1[r] = out[base + col1];
    }

#pragma unroll 1
    for (int t = 0; t < TT; ++t) {
        const __bf16* hprev = hb + (size_t)(t & 1) * NB * HH;
        __bf16*       hnext = hb + (size_t)((t + 1) & 1) * NB * HH;

        v8f acc0 = nxt0, acc1 = nxt1;   // C init = xw(t)

        // Pipelined load of xw(t+1): independent, overlaps the WMMA chain.
        if (t + 1 < TT) {
#pragma unroll
            for (int r = 0; r < 8; ++r) {
                const int row = m0 + r + (hs ? 8 : 0);
                const size_t base = ((size_t)row * TT + (t + 1)) * HH;
                nxt0[r] = out[base + col0];
                nxt1[r] = out[base + col1];
            }
        }

        // h_prev @ Wh via 8 chained bf16 WMMAs per tile (2 tiles interleaved)
#pragma unroll
        for (int k0 = 0; k0 < HH; k0 += 32) {
            const int kbase = k0 + (hs ? 16 : 0);
            v8bf lo = *(const v8bf*)(hprev + arow * HH + kbase);
            v8bf hi = *(const v8bf*)(hprev + arow * HH + kbase + 8);
            v16bf a;
#pragma unroll
            for (int i = 0; i < 8; ++i) { a[i] = lo[i]; a[8 + i] = hi[i]; }

            const int kk = k0 + (hs ? 16 : 0);
            v16bf b0 = *(const v16bf*)(whs + (size_t)col0 * HH + kk);
            v16bf b1 = *(const v16bf*)(whs + (size_t)col1 * HH + kk);
            acc0 = __builtin_amdgcn_wmma_f32_16x16x32_bf16(
                false, a, false, b0, (short)0, acc0, false, false);
            acc1 = __builtin_amdgcn_wmma_f32_16x16x32_bf16(
                false, a, false, b1, (short)0, acc1, false, false);
        }

        // tanh, write fp32 result in place, stage bf16 h into next buffer
#pragma unroll
        for (int r = 0; r < 8; ++r) {
            const int row = m0 + r + (hs ? 8 : 0);
            const size_t base = ((size_t)row * TT + t) * HH;
            const float h0v = tanhf(acc0[r]);
            const float h1v = tanhf(acc1[r]);
            out[base + col0] = h0v;
            out[base + col1] = h1v;
            hnext[row * HH + col0] = f2bf(h0v);
            hnext[row * HH + col1] = f2bf(h1v);
        }
        __syncthreads();
    }
}

// ---------------------------------------------------------------------------
extern "C" void kernel_launch(void* const* d_in, const int* in_sizes, int n_in,
                              void* d_out, int out_size, void* d_ws, size_t ws_size,
                              hipStream_t stream) {
    const float* x  = (const float*)d_in[0];   // [N,T,D]
    const float* h0 = (const float*)d_in[1];   // [N,H]
    const float* Wx = (const float*)d_in[2];   // [D,H]
    const float* Wh = (const float*)d_in[3];   // [H,H]
    const float* b  = (const float*)d_in[4];   // [H]
    float* out = (float*)d_out;                // [N,T,H]

    unsigned short* WxT = (unsigned short*)d_ws;            // 128KB bf16
    unsigned short* WhT = WxT + (size_t)DD * HH;            // 128KB bf16

    // 0) weights -> transposed bf16 in workspace
    prep_weights_kernel<<<(DD * HH + 255) / 256, 256, 0, stream>>>(Wx, Wh, WxT, WhT);

    // 1) xW = x @ Wx + b  -> d_out   (M=65536 rows / 128 per block, H/64 cols)
    dim3 ggrid((NB * TT) / 128, HH / 64);
    xw_gemm_kernel<<<ggrid, 256, 0, stream>>>(x, WxT, b, out);

    // 2) sequential scan: one WGP, weights + h state resident in LDS
    const size_t shmem = (size_t)HH * HH * 2 + (size_t)2 * NB * HH * 2; // 192KB
    rnn_scan_kernel<<<1, 1024, shmem, stream>>>(h0, WhT, out);
}
